// ExternalMemoryRetriever_21139829031184
// MI455X (gfx1250) — matrix operand order repaired and checked by reading, not compiled, not run on hardware
//
#include <hip/hip_runtime.h>
#include <hip/hip_bf16.h>

// ---------------------------------------------------------------------------
// Types for CDNA5 WMMA (wave32): v16bf A/B fragments, v8f accumulator.
// ---------------------------------------------------------------------------
typedef __attribute__((ext_vector_type(16))) __bf16 v16bf;
typedef __attribute__((ext_vector_type(8)))  float  v8f;
typedef __attribute__((ext_vector_type(4)))  unsigned int v4u;
typedef __attribute__((ext_vector_type(8)))  unsigned int v8u;

union Frag {
    v16bf v;
    uint4 u[2];
};

__device__ __forceinline__ unsigned int f2bf_rne(float f) {
    unsigned int x = __float_as_uint(f);
    unsigned int r = x + 0x7FFFu + ((x >> 16) & 1u);
    return r >> 16;
}
__device__ __forceinline__ unsigned int pack_bf16(float a, float b) {
    return f2bf_rne(a) | (f2bf_rne(b) << 16);
}

#define ALPHA 0.8f
#define D_DIM 768
#define N_MEM 4096
#define Q_TOK 32
#define B_SZ  16
#define QQ    32
#define TOPK  9

// LDS layout for local_sim_wmma (in u32 units)
#define RAW_U32   (Q_TOK * D_DIM)          // 24576 u32 : raw f32 tile from TDM
#define BF_STRIDE 388                      // padded row stride (packed bf16 pairs)
#define BF_U32    (Q_TOK * BF_STRIDE)      // 12416 u32 : converted bf16 tile
#define LDS_U32   (RAW_U32 + BF_U32)       // 36992 u32 = 147968 B < 320 KB

// ---------------------------------------------------------------------------
// K0: convert f32 rows -> packed bf16 (2 per uint), row-major.
// ---------------------------------------------------------------------------
__global__ void cvt_f32_to_bf16pk(const float* __restrict__ x,
                                  unsigned int* __restrict__ out,
                                  int n_pairs) {
    int i = blockIdx.x * blockDim.x + threadIdx.x;
    if (i < n_pairs) {
        float2 f = ((const float2*)x)[i];
        out[i] = pack_bf16(f.x, f.y);
    }
}

// ---------------------------------------------------------------------------
// K1: reciprocal L2 norms, one wave32 per row of length D=768.
// ---------------------------------------------------------------------------
__global__ void rnorm_rows(const float* __restrict__ x,
                           float* __restrict__ r, int rows) {
    int wave = (blockIdx.x * blockDim.x + threadIdx.x) >> 5;
    int lane = threadIdx.x & 31;
    if (wave >= rows) return;
    const float* p = x + (size_t)wave * D_DIM;
    float s = 0.f;
    #pragma unroll
    for (int i = 0; i < D_DIM / 32; ++i) {
        float v = p[lane + i * 32];
        s += v * v;
    }
    #pragma unroll
    for (int off = 16; off > 0; off >>= 1) s += __shfl_xor(s, off, 32);
    if (lane == 0) {
        float n = sqrtf(s);
        r[wave] = 1.0f / fmaxf(n, 1e-12f);
    }
}

// ---------------------------------------------------------------------------
// K2: pooled scores via WMMA bf16: scores[b,n] = ALPHA * <qf_n[b], mi_n[n]>.
// ---------------------------------------------------------------------------
__global__ void pooled_scores_wmma(const unsigned int* __restrict__ qf_bf,  // [16][384]
                                   const float* __restrict__ mi,            // [4096][768]
                                   const float* __restrict__ r_qf,          // [16]
                                   const float* __restrict__ r_mi,          // [4096]
                                   float* __restrict__ scores) {            // [16][4096]
    int wid  = (blockIdx.x * blockDim.x + threadIdx.x) >> 5;
    int lane = threadIdx.x & 31;
    int n0   = wid * 16;
    if (n0 >= N_MEM) return;

    int col  = lane & 15;
    int half = lane >> 4;

    v8f acc = {};
    const unsigned int* arow = qf_bf + (size_t)col * (D_DIM / 2);
    const float*        brow = mi + (size_t)(n0 + col) * D_DIM;

    for (int d = 0; d < D_DIM; d += 32) {
        // prefetch the B row ahead (global_prefetch_b8)
        if (d + 128 < D_DIM) __builtin_prefetch(brow + d + 128, 0, 1);

        // A fragment: V0-3 = K[d + half*8 ..), V4-7 = K[d+16 + half*8 ..)
        Frag a;
        a.u[0] = *(const uint4*)(arow + (d >> 1) + half * 4);
        a.u[1] = *(const uint4*)(arow + (d >> 1) + 8 + half * 4);
        // B fragment: V0-3 = K[d + half*8 ..), V4-7 = K[d+16 + half*8 ..)
        const float* bp = brow + d + half * 8;
        unsigned int us[8];
        #pragma unroll
        for (int j = 0; j < 4; ++j) {
            float4 f = ((const float4*)bp)[j < 2 ? j : j + 2];
            us[2 * j]     = pack_bf16(f.x, f.y);
            us[2 * j + 1] = pack_bf16(f.z, f.w);
        }
        Frag b;
        b.u[0] = make_uint4(us[0], us[1], us[2], us[3]);
        b.u[1] = make_uint4(us[4], us[5], us[6], us[7]);

        acc = __builtin_amdgcn_wmma_f32_16x16x32_bf16(false, a.v, false, b.v,
                                                      (short)0, acc, false, false);
    }

    int   ncol = n0 + col;
    float rm   = r_mi[ncol];
    #pragma unroll
    for (int v = 0; v < 8; ++v) {
        int m = v + half * 8;                       // batch row 0..15
        scores[(size_t)m * N_MEM + ncol] = ALPHA * acc[v] * r_qf[m] * rm;
    }
}

// ---------------------------------------------------------------------------
// K3: token-local similarity. One block per memory item n, 8 waves.
// The item's full 32x768 f32 token tile (96 KB) is DMA'd into LDS by the
// Tensor Data Mover (one tensor_load_to_lds per workgroup, TENSORcnt-
// tracked), converted once to packed bf16 in LDS, then the whole K loop runs
// barrier-free on ds_load_b128 fragments + v_wmma_f32_16x16x32_bf16.
// ---------------------------------------------------------------------------
__global__ void __launch_bounds__(256)
local_sim_wmma(const unsigned int* __restrict__ qt_bf,   // [512][384]
               const float* __restrict__ memtok,         // [4096][32][768]
               const float* __restrict__ r_qt,           // [512]
               const float* __restrict__ r_mt,           // [4096*32]
               float* __restrict__ scores) {             // [16][4096]
    __shared__ unsigned int LBUF[LDS_U32];   // [0,24576): raw f32, [24576,..): bf16

    int n    = blockIdx.x;
    int t    = threadIdx.x;
    int wave = t >> 5;
    int lane = t & 31;
    int col  = lane & 15;
    int half = lane >> 4;
    int m0   = wave * 64;

    // ---- TDM: DMA the whole 32x768 f32 tile for item n into LDS ----------
    if (t < 32) {
        unsigned long long gaddr =
            (unsigned long long)(uintptr_t)(memtok + (size_t)n * Q_TOK * D_DIM);
        unsigned int lds_base = (unsigned int)(uintptr_t)&LBUF[0];

        v4u g0;
        g0[0] = 1u;                                   // count=1, user descriptor
        g0[1] = lds_base;                             // lds_addr (bytes)
        g0[2] = (unsigned int)(gaddr & 0xFFFFFFFFull);            // global_addr lo
        g0[3] = (unsigned int)((gaddr >> 32) & 0x1FFFFFFull)      // global_addr hi
              | (2u << 30);                           // type = 2 (image)

        v8u g1;
        g1[0] = (2u << 16);                           // data_size = 4 bytes
        g1[1] = ((unsigned int)D_DIM << 16);          // tensor_dim0[15:0] @ bits 63:48
        g1[2] = ((unsigned int)Q_TOK << 16);          // tensor_dim1[15:0] @ bits 95:80; dim0 hi = 0
        g1[3] = ((unsigned int)D_DIM << 16);          // tile_dim0 @ bits 127:112 (dim1 hi = 0)
        g1[4] = (unsigned int)Q_TOK;                  // tile_dim1 @ bits 143:128
        g1[5] = (unsigned int)D_DIM;                  // tensor_dim0_stride[31:0]
        g1[6] = 0u;                                   // stride hi / dim1_stride lo
        g1[7] = 0u;

        asm volatile("tensor_load_to_lds %0, %1"
                     :
                     : "s"(g0), "s"(g1)
                     : "memory");
        __builtin_amdgcn_s_wait_tensorcnt(0);
    }
    __syncthreads();

    // ---- one-shot f32 -> packed-bf16 conversion pass in LDS ---------------
    // 12288 packed uints total, 48 per thread.
    #pragma unroll 4
    for (int i = 0; i < 48; ++i) {
        int p   = t + i * 256;           // packed-pair index
        int row = p / 384;               // token row (768 f32 = 384 pairs)
        int cp  = p - row * 384;         // pair column within row
        uint2 w = *(const uint2*)&LBUF[2 * p];
        LBUF[RAW_U32 + row * BF_STRIDE + cp] =
            pack_bf16(__uint_as_float(w.x), __uint_as_float(w.y));
    }
    __syncthreads();

    // ---- K loop: barrier-free WMMA over D ---------------------------------
    v8f acc[4][2];
    #pragma unroll
    for (int i = 0; i < 4; ++i)
        #pragma unroll
        for (int j = 0; j < 2; ++j) acc[i][j] = (v8f){};

    const unsigned int* b0base = &LBUF[RAW_U32 + col * BF_STRIDE];
    const unsigned int* b1base = &LBUF[RAW_U32 + (col + 16) * BF_STRIDE];

    for (int d = 0; d < D_DIM; d += 32) {
        int ko = (d >> 1) + half * 8;    // 16 K-values per lane, half-selected
        Frag b0, b1;
        b0.u[0] = *(const uint4*)(b0base + ko);
        b0.u[1] = *(const uint4*)(b0base + ko + 4);
        b1.u[0] = *(const uint4*)(b1base + ko);
        b1.u[1] = *(const uint4*)(b1base + ko + 4);

        #pragma unroll
        for (int mi_ = 0; mi_ < 4; ++mi_) {
            const unsigned int* arow =
                qt_bf + (size_t)(m0 + mi_ * 16 + col) * (D_DIM / 2);
            Frag a;
            a.u[0] = *(const uint4*)(arow + (d >> 1) + half * 8);
            a.u[1] = *(const uint4*)(arow + (d >> 1) + half * 8 + 4);

            acc[mi_][0] = __builtin_amdgcn_wmma_f32_16x16x32_bf16(
                false, a.v, false, b0.v, (short)0, acc[mi_][0], false, false);
            acc[mi_][1] = __builtin_amdgcn_wmma_f32_16x16x32_bf16(
                false, a.v, false, b1.v, (short)0, acc[mi_][1], false, false);
        }
    }

    // ---- epilogue: normalize + max over (32 q-rows x 32 token-cols) -------
    const float* rmp = r_mt + (size_t)n * Q_TOK;
    float rm0 = rmp[col];
    float rm1 = rmp[col + 16];

    float bmax[2] = {-__builtin_inff(), -__builtin_inff()};
    #pragma unroll
    for (int mi_ = 0; mi_ < 4; ++mi_) {
        int mbase = m0 + mi_ * 16 + half * 8;
        #pragma unroll
        for (int v = 0; v < 8; ++v) {
            float rq = r_qt[mbase + v];
            float s0 = acc[mi_][0][v] * rq * rm0;
            float s1 = acc[mi_][1][v] * rq * rm1;
            float s  = fmaxf(s0, s1);
            bmax[mi_ >> 1] = fmaxf(bmax[mi_ >> 1], s);
        }
    }
    #pragma unroll
    for (int j = 0; j < 2; ++j) {
        float m = bmax[j];
        #pragma unroll
        for (int off = 16; off > 0; off >>= 1) m = fmaxf(m, __shfl_xor(m, off, 32));
        if (lane == 0) {
            int b = wave * 2 + j;
            scores[(size_t)b * N_MEM + n] += (1.0f - ALPHA) * m;
        }
    }
}

// ---------------------------------------------------------------------------
// K4: top-9 per batch row (iterative argmax, lowest-index tie-break).
// out[0:144) = values, out[144:288) = indices (as float).
// ---------------------------------------------------------------------------
__global__ void topk9_kernel(const float* __restrict__ scores,
                             float* __restrict__ out) {
    __shared__ float s[N_MEM];
    __shared__ float rv[256];
    __shared__ int   ri[256];
    int b = blockIdx.x;
    int t = threadIdx.x;
    for (int i = t; i < N_MEM; i += 256) s[i] = scores[(size_t)b * N_MEM + i];
    __syncthreads();

    for (int j = 0; j < TOPK; ++j) {
        float bv = -__builtin_inff();
        int   bi = N_MEM;
        for (int i = t; i < N_MEM; i += 256) {
            float v = s[i];
            if (v > bv) { bv = v; bi = i; }
        }
        rv[t] = bv; ri[t] = bi;
        __syncthreads();
        for (int off = 128; off > 0; off >>= 1) {
            if (t < off) {
                float ov = rv[t + off]; int oi = ri[t + off];
                if (ov > rv[t] || (ov == rv[t] && oi < ri[t])) { rv[t] = ov; ri[t] = oi; }
            }
            __syncthreads();
        }
        if (t == 0) {
            out[b * TOPK + j]               = rv[0];
            out[B_SZ * TOPK + b * TOPK + j] = (float)ri[0];
            s[ri[0]] = -__builtin_inff();
        }
        __syncthreads();
    }
}

// ---------------------------------------------------------------------------
// Host-side launcher
// ---------------------------------------------------------------------------
extern "C" void kernel_launch(void* const* d_in, const int* in_sizes, int n_in,
                              void* d_out, int out_size, void* d_ws, size_t ws_size,
                              hipStream_t stream) {
    const float* qf   = (const float*)d_in[0];   // [16][768]
    const float* qt   = (const float*)d_in[1];   // [16][32][768]
    const float* mi   = (const float*)d_in[2];   // [4096][768]
    const float* mtok = (const float*)d_in[3];   // [4096][32][768]
    (void)in_sizes; (void)n_in; (void)out_size; (void)ws_size;

    unsigned int* qt_bf = (unsigned int*)d_ws;                       // 512*384 u32
    unsigned int* qf_bf = qt_bf + (size_t)512 * 384;                 // 16*384  u32
    float* r_qt   = (float*)(qf_bf + (size_t)16 * 384);              // 512
    float* r_qf   = r_qt + 512;                                      // 16
    float* r_mi   = r_qf + 16;                                       // 4096
    float* r_mt   = r_mi + 4096;                                     // 131072
    float* scores = r_mt + 131072;                                   // 16*4096
    float* out    = (float*)d_out;

    {
        int pairs_qt = (QQ * B_SZ * D_DIM) / 2;      // 196608
        cvt_f32_to_bf16pk<<<dim3((pairs_qt + 255) / 256), dim3(256), 0, stream>>>(qt, qt_bf, pairs_qt);
        int pairs_qf = (B_SZ * D_DIM) / 2;           // 6144
        cvt_f32_to_bf16pk<<<dim3((pairs_qf + 255) / 256), dim3(256), 0, stream>>>(qf, qf_bf, pairs_qf);
    }

    rnorm_rows<<<dim3((512 + 7) / 8), dim3(256), 0, stream>>>(qt,   r_qt, 512);
    rnorm_rows<<<dim3(2),             dim3(256), 0, stream>>>(qf,   r_qf, 16);
    rnorm_rows<<<dim3(4096 / 8),      dim3(256), 0, stream>>>(mi,   r_mi, 4096);
    rnorm_rows<<<dim3(131072 / 8),    dim3(256), 0, stream>>>(mtok, r_mt, 131072);

    pooled_scores_wmma<<<dim3(32), dim3(256), 0, stream>>>(qf_bf, mi, r_qf, r_mi, scores);

    local_sim_wmma<<<dim3(N_MEM), dim3(256), 0, stream>>>(qt_bf, mtok, r_qt, r_mt, scores);

    topk9_kernel<<<dim3(B_SZ), dim3(256), 0, stream>>>(scores, out);
}
